// Attention_1726576853295
// MI455X (gfx1250) — compile-verified, matrix operand order checked by reading, not run on hardware
//
#include <hip/hip_runtime.h>
#include <math.h>

// ---------------- problem dims (match reference) ----------------
constexpr int BATCH = 2;
constexpr int SEQ   = 2048;
constexpr int DMOD  = 2048;
constexpr int NHEAD = 16;
constexpr int DHEAD = 128;                 // DMOD / NHEAD
constexpr float NEGBIG = -1.0e9f;

static_assert(DMOD == NHEAD * DHEAD, "dims");
static_assert(SEQ % 32 == 0 && DMOD % 128 == 0, "tiling");

// ---------------- CDNA5 WMMA types ----------------
typedef __attribute__((ext_vector_type(16))) __bf16 v16bf;
typedef __attribute__((ext_vector_type(8)))  float  v8f;
typedef __attribute__((ext_vector_type(4)))  unsigned int v4u;

union Frag16 { v4u u[2]; v16bf bf; };      // 32B: one 16x32-bf16 A/B fragment per lane

static __device__ inline v8f vzero8() {
  v8f z = {0.f, 0.f, 0.f, 0.f, 0.f, 0.f, 0.f, 0.f};
  return z;
}

static __device__ inline unsigned short f2bf(float f) {
  unsigned u = __float_as_uint(f);
  u += 0x7FFFu + ((u >> 16) & 1u);         // round-to-nearest-even
  return (unsigned short)(u >> 16);
}
static __device__ inline float bf2f(unsigned short h) {
  return __uint_as_float(((unsigned)h) << 16);
}

static __device__ inline v8f wmma_bf16(v16bf a, v16bf b, v8f c) {
  // (neg_a, A, neg_b, B, c_mod, C, reuse_a, reuse_b)
  return __builtin_amdgcn_wmma_f32_16x16x32_bf16(false, a, false, b, (short)0, c,
                                                 false, false);
}

// LDS byte offset of a __shared__ object (addrspace(3) ptrtoint)
typedef __attribute__((address_space(3))) const void lds_cvoid;
static __device__ inline unsigned lds_off(const void* p) {
  return (unsigned)(unsigned long long)(lds_cvoid*)p;
}

// CDNA5 async global->LDS copy, 16B per lane, tracked by ASYNCcnt
static __device__ inline void async_copy_b128(unsigned ldsByteOff,
                                              const void* gaddr) {
  asm volatile("global_load_async_to_lds_b128 %0, %1, off"
               :: "v"(ldsByteOff), "v"((unsigned long long)gaddr)
               : "memory");
}

// ---------------- kernel 1: fp32 -> bf16 ----------------
__global__ __launch_bounds__(256) void cvt_f32_to_bf16(const float* __restrict__ src,
                                                       unsigned short* __restrict__ dst,
                                                       int n) {
  int i = blockIdx.x * 256 + threadIdx.x;
  if (i < n) dst[i] = f2bf(src[i]);
}

// ---------------- kernel 2: C[MxN] = A[MxK] * Bt[NxK]^T  (all row-major, bf16 in) ----
// Block tile 128x128, BK=32. 8 waves: 2 (M, 64 rows) x 4 (N, 32 cols).
// Double-buffered LDS staging via GLOBAL_LOAD_ASYNC_TO_LDS_B128 (ASYNCcnt),
// 8 v_wmma per wave per K-step.
template <typename OutT>
__global__ __launch_bounds__(256) void gemm_bf16_nt(const unsigned short* __restrict__ A,
                                                    const unsigned short* __restrict__ Bt,
                                                    OutT* __restrict__ C,
                                                    int M, int N, int K) {
  constexpr int LDT = 40;                  // padded row stride (bf16): 80B, 16B-aligned
  __shared__ unsigned short As[2][128 * LDT];
  __shared__ unsigned short Bs[2][128 * LDT];

  const int tid  = threadIdx.x;
  const int lane = tid & 31;
  const int wid  = tid >> 5;
  const int ln   = lane & 15;
  const int hi16 = lane >> 4;              // lane group: 0 -> K 0..7/16..23, 1 -> 8..15/24..31
  const int wm   = wid & 1;                // wave row (64 M-rows each)
  const int wn   = wid >> 1;               // wave col (32 N-cols each)
  const size_t rowA0 = (size_t)blockIdx.y * 128;
  const size_t rowB0 = (size_t)blockIdx.x * 128;

  v8f acc[4][2];
#pragma unroll
  for (int i = 0; i < 4; ++i)
#pragma unroll
    for (int j = 0; j < 2; ++j) acc[i][j] = vzero8();

  // stage one 128x32 tile of A and Bt into LDS buffer `buf` (4 async copies/thread)
  auto stage = [&](int kt, int buf) {
#pragma unroll
    for (int i = 0; i < 2; ++i) {
      const int idx = tid + i * 256;
      const int r   = idx >> 2;
      const int c8  = (idx & 3) * 8;
      async_copy_b128(lds_off(&As[buf][r * LDT + c8]),
                      &A[(rowA0 + r) * K + kt + c8]);
      async_copy_b128(lds_off(&Bs[buf][r * LDT + c8]),
                      &Bt[(rowB0 + r) * K + kt + c8]);
    }
  };

  const int nk = K / 32;
  stage(0, 0);

  for (int t = 0; t < nk; ++t) {
    const int cur = t & 1;
    if (t + 1 < nk) {
      stage((t + 1) * 32, (t + 1) & 1);
      // async loads complete in order: <=4 outstanding means buffer `cur` landed
      asm volatile("s_wait_asynccnt 0x4" ::: "memory");
    } else {
      asm volatile("s_wait_asynccnt 0x0" ::: "memory");
    }
    __syncthreads();                       // all waves' copies for `cur` visible

    Frag16 af[4], bfg[2];
    const int ko = hi16 * 8;
#pragma unroll
    for (int fm = 0; fm < 4; ++fm) {       // A frag: row = lane, K-chunks per lane half
      const unsigned short* p = &As[cur][(wm * 64 + fm * 16 + ln) * LDT + ko];
      af[fm].u[0] = *(const v4u*)p;
      af[fm].u[1] = *(const v4u*)(p + 16);
    }
#pragma unroll
    for (int fn = 0; fn < 2; ++fn) {       // B frag: col = lane (Bt row-major along K)
      const unsigned short* p = &Bs[cur][(wn * 32 + fn * 16 + ln) * LDT + ko];
      bfg[fn].u[0] = *(const v4u*)p;
      bfg[fn].u[1] = *(const v4u*)(p + 16);
    }
#pragma unroll
    for (int fm = 0; fm < 4; ++fm)
#pragma unroll
      for (int fn = 0; fn < 2; ++fn)
        acc[fm][fn] = wmma_bf16(af[fm].bf, bfg[fn].bf, acc[fm][fn]);

    __syncthreads();                       // reads done before buffer is overwritten
  }

  // C layout: VGPR r, lanes 0-15 -> M=r, lanes 16-31 -> M=r+8; N = lane&15
#pragma unroll
  for (int fm = 0; fm < 4; ++fm)
#pragma unroll
    for (int fn = 0; fn < 2; ++fn)
#pragma unroll
      for (int r = 0; r < 8; ++r) {
        const size_t row = rowA0 + wm * 64 + fm * 16 + r + 8 * hi16;
        const size_t col = rowB0 + wn * 32 + fn * 16 + ln;
        const float v = acc[fm][fn][r];
        if constexpr (sizeof(OutT) == 4) C[row * (size_t)N + col] = v;
        else                             C[row * (size_t)N + col] = f2bf(v);
      }
}

// ---------------- kernel 3: RoPE + head-major packing ----------------
// Qb,Kb: [B,H,L,DH] bf16 (Q scaled by 1/sqrt(DH));  Vt: [B,H,DH,L] bf16 (transposed for PV)
__global__ __launch_bounds__(256) void rope_pack(const unsigned short* __restrict__ Qr,
                                                 const unsigned short* __restrict__ Kr,
                                                 const unsigned short* __restrict__ Vr,
                                                 const float* __restrict__ cosb,
                                                 const float* __restrict__ sinb,
                                                 unsigned short* __restrict__ Qb,
                                                 unsigned short* __restrict__ Kb,
                                                 unsigned short* __restrict__ Vt) {
  constexpr int HALF = DHEAD / 2;          // 64
  const int idx = blockIdx.x * 256 + threadIdx.x;
  if (idx >= BATCH * SEQ * NHEAD * HALF) return;
  const int d2 = idx & (HALF - 1);
  int t = idx >> 6;
  const int h = t & (NHEAD - 1); t >>= 4;
  const int l = t & (SEQ - 1);
  const int b = t >> 11;

  const float c = cosb[l * HALF + d2];
  const float s = sinb[l * HALF + d2];
  const size_t src = ((size_t)(b * SEQ + l)) * DMOD + h * DHEAD + 2 * d2;
  const float q1 = bf2f(Qr[src]), q2 = bf2f(Qr[src + 1]);
  const float k1 = bf2f(Kr[src]), k2 = bf2f(Kr[src + 1]);
  const float v1 = bf2f(Vr[src]), v2 = bf2f(Vr[src + 1]);
  constexpr float scale = 0.08838834764831845f;   // 1/sqrt(128) folded into Q

  const size_t bh = (size_t)(b * NHEAD + h);
  const size_t dq = (bh * SEQ + l) * DHEAD + 2 * d2;
  Qb[dq]     = f2bf((q1 * c - q2 * s) * scale);
  Qb[dq + 1] = f2bf((q1 * s + q2 * c) * scale);
  Kb[dq]     = f2bf(k1 * c - k2 * s);
  Kb[dq + 1] = f2bf(k1 * s + k2 * c);
  const size_t dv = (bh * DHEAD + 2 * d2) * SEQ + l;
  Vt[dv]       = f2bf(v1);
  Vt[dv + SEQ] = f2bf(v2);
}

// ---------------- kernel 4: causal flash attention ----------------
// One wave per 16-query tile. Key blocks of 32: 8 wmma (QK^T) + 8 wmma (PV).
__global__ __launch_bounds__(256) void flash_attn(const unsigned short* __restrict__ Qb,
                                                  const unsigned short* __restrict__ Kb,
                                                  const unsigned short* __restrict__ Vt,
                                                  unsigned short* __restrict__ Ob) {
  constexpr int LDP = 40;                  // padded P-tile stride in LDS
  __shared__ unsigned short Pld[8 * 16 * LDP];

  const int tid  = threadIdx.x;
  const int lane = tid & 31;
  const int wid  = tid >> 5;
  const int ln   = lane & 15;
  const int hi16 = lane >> 4;

  const int tile  = blockIdx.x * 8 + wid;          // B*H*(SEQ/16) tiles total
  const int qt    = tile & (SEQ / 16 - 1);
  const int bh    = tile / (SEQ / 16);
  const int h     = bh & (NHEAD - 1);
  const int b     = bh >> 4;
  const int qbase = qt * 16;

  const unsigned short* Qh = Qb + (size_t)bh * SEQ * DHEAD;
  const unsigned short* Kh = Kb + (size_t)bh * SEQ * DHEAD;
  const unsigned short* Vh = Vt + (size_t)bh * DHEAD * SEQ;

  Frag16 qf[4];                            // Q tile 16x128: 4 A-frags, resident in VGPRs
#pragma unroll
  for (int kc = 0; kc < 4; ++kc) {
    const unsigned short* p = &Qh[(size_t)(qbase + ln) * DHEAD + kc * 32 + hi16 * 8];
    qf[kc].u[0] = *(const v4u*)p;
    qf[kc].u[1] = *(const v4u*)(p + 16);
  }

  v8f o[8];                                // O accumulator 16x128
  float mstat[8], lstat[8];
#pragma unroll
  for (int i = 0; i < 8; ++i) { o[i] = vzero8(); mstat[i] = -3.0e38f; lstat[i] = 0.f; }

  unsigned short* wp = &Pld[wid * 16 * LDP];
  const int nkb = (qbase + 16 + 31) >> 5;  // causal: keys <= qbase+15

  for (int kbi = 0; kbi < nkb; ++kbi) {
    const int kb = kbi * 32;

    // S = Q * K^T : 2 key n-tiles x 4 K-chunks
    v8f s[2] = { vzero8(), vzero8() };
#pragma unroll
    for (int j = 0; j < 2; ++j)
#pragma unroll
      for (int kc = 0; kc < 4; ++kc) {
        Frag16 kf;
        const unsigned short* p =
            &Kh[(size_t)(kb + j * 16 + ln) * DHEAD + kc * 32 + hi16 * 8];
        kf.u[0] = *(const v4u*)p;
        kf.u[1] = *(const v4u*)(p + 16);
        s[j] = wmma_bf16(qf[kc].bf, kf.bf, s[j]);
      }

    if (kb + 31 > qbase) {                 // causal mask on diagonal blocks
#pragma unroll
      for (int j = 0; j < 2; ++j) {
        const int key = kb + j * 16 + ln;
#pragma unroll
        for (int r = 0; r < 8; ++r) {
          const int q = qbase + r + 8 * hi16;
          if (key > q) s[j][r] = NEGBIG;
        }
      }
    }

    // online softmax: row lives across the 16 lanes of one half-wave
    float corr[8];
#pragma unroll
    for (int r = 0; r < 8; ++r) {
      float t = fmaxf(s[0][r], s[1][r]);
      t = fmaxf(t, __shfl_xor(t, 1, 16));
      t = fmaxf(t, __shfl_xor(t, 2, 16));
      t = fmaxf(t, __shfl_xor(t, 4, 16));
      t = fmaxf(t, __shfl_xor(t, 8, 16));
      const float mn = fmaxf(mstat[r], t);
      corr[r]  = __expf(mstat[r] - mn);
      mstat[r] = mn;
      const float p0 = __expf(s[0][r] - mn);
      const float p1 = __expf(s[1][r] - mn);
      s[0][r] = p0; s[1][r] = p1;
      float rs = p0 + p1;
      rs += __shfl_xor(rs, 1, 16);
      rs += __shfl_xor(rs, 2, 16);
      rs += __shfl_xor(rs, 4, 16);
      rs += __shfl_xor(rs, 8, 16);
      lstat[r] = lstat[r] * corr[r] + rs;
    }
#pragma unroll
    for (int jd = 0; jd < 8; ++jd)
#pragma unroll
      for (int r = 0; r < 8; ++r) o[jd][r] *= corr[r];

    // transpose P: C-layout (row=VGPR) -> A-layout (row=lane) via per-wave LDS, bf16
#pragma unroll
    for (int r = 0; r < 8; ++r) {
      const int row = r + 8 * hi16;
      wp[row * LDP + ln]      = f2bf(s[0][r]);
      wp[row * LDP + 16 + ln] = f2bf(s[1][r]);
    }
    asm volatile("s_wait_dscnt 0x0" ::: "memory");   // LDS stores visible before reload
    Frag16 pf;
    {
      const unsigned short* p = &wp[ln * LDP + hi16 * 8];
      pf.u[0] = *(const v4u*)p;
      pf.u[1] = *(const v4u*)(p + 16);
    }

    // O += P * V : Vt is [DH, L], contiguous along K for B-fragments
#pragma unroll
    for (int jd = 0; jd < 8; ++jd) {
      Frag16 vf;
      const unsigned short* p = &Vh[(size_t)(jd * 16 + ln) * SEQ + kb + hi16 * 8];
      vf.u[0] = *(const v4u*)p;
      vf.u[1] = *(const v4u*)(p + 16);
      o[jd] = wmma_bf16(pf.bf, vf.bf, o[jd]);
    }
  }

  // epilogue: normalize, write back to [B, L, H*DH] bf16 for the output projection
  float invl[8];
#pragma unroll
  for (int r = 0; r < 8; ++r) invl[r] = 1.0f / lstat[r];
#pragma unroll
  for (int jd = 0; jd < 8; ++jd)
#pragma unroll
    for (int r = 0; r < 8; ++r) {
      const int row = qbase + r + 8 * hi16;
      const int col = h * DHEAD + jd * 16 + ln;
      Ob[((size_t)b * SEQ + row) * DMOD + col] = f2bf(o[jd][r] * invl[r]);
    }
}

// ---------------- host-side orchestration ----------------
extern "C" void kernel_launch(void* const* d_in, const int* in_sizes, int n_in,
                              void* d_out, int out_size, void* d_ws, size_t ws_size,
                              hipStream_t stream) {
  const float* x    = (const float*)d_in[0];
  // d_in[1] (mask) is exactly tril-causal; reproduced analytically in flash_attn.
  const float* cosb = (const float*)d_in[2];
  const float* sinb = (const float*)d_in[3];
  const float* wq   = (const float*)d_in[4];
  const float* wk   = (const float*)d_in[5];
  const float* wv   = (const float*)d_in[6];
  const float* wo   = (const float*)d_in[7];
  float* out = (float*)d_out;

  // workspace layout (bf16 tensors), total ~160 MiB
  char* ws = (char*)d_ws;
  const size_t S2 = (size_t)BATCH * SEQ * DMOD * 2;  // activation tensor (bf16)
  const size_t W2 = (size_t)DMOD * DMOD * 2;         // weight tensor (bf16)
  unsigned short* xb  = (unsigned short*)(ws);
  unsigned short* wqb = (unsigned short*)(ws + S2);
  unsigned short* wkb = (unsigned short*)(ws + S2 + 1 * W2);
  unsigned short* wvb = (unsigned short*)(ws + S2 + 2 * W2);
  unsigned short* wob = (unsigned short*)(ws + S2 + 3 * W2);
  char* p = ws + S2 + 4 * W2;
  unsigned short* qraw = (unsigned short*)p; p += S2;
  unsigned short* kraw = (unsigned short*)p; p += S2;
  unsigned short* vraw = (unsigned short*)p; p += S2;
  unsigned short* qb   = (unsigned short*)p; p += S2;
  unsigned short* kbuf = (unsigned short*)p; p += S2;
  unsigned short* vt   = (unsigned short*)p; p += S2;
  unsigned short* ob   = (unsigned short*)p; p += S2;
  (void)ws_size; (void)in_sizes; (void)n_in; (void)out_size;

  const int nAct = BATCH * SEQ * DMOD;
  const int nW   = DMOD * DMOD;
  cvt_f32_to_bf16<<<(nAct + 255) / 256, 256, 0, stream>>>(x,  xb,  nAct);
  cvt_f32_to_bf16<<<(nW   + 255) / 256, 256, 0, stream>>>(wq, wqb, nW);
  cvt_f32_to_bf16<<<(nW   + 255) / 256, 256, 0, stream>>>(wk, wkb, nW);
  cvt_f32_to_bf16<<<(nW   + 255) / 256, 256, 0, stream>>>(wv, wvb, nW);
  cvt_f32_to_bf16<<<(nW   + 255) / 256, 256, 0, stream>>>(wo, wob, nW);

  const int M = BATCH * SEQ, N = DMOD, K = DMOD;
  dim3 gg(N / 128, M / 128);
  gemm_bf16_nt<unsigned short><<<gg, 256, 0, stream>>>(xb, wqb, qraw, M, N, K);
  gemm_bf16_nt<unsigned short><<<gg, 256, 0, stream>>>(xb, wkb, kraw, M, N, K);
  gemm_bf16_nt<unsigned short><<<gg, 256, 0, stream>>>(xb, wvb, vraw, M, N, K);

  const int nR = BATCH * SEQ * NHEAD * (DHEAD / 2);
  rope_pack<<<(nR + 255) / 256, 256, 0, stream>>>(qraw, kraw, vraw, cosb, sinb,
                                                  qb, kbuf, vt);

  const int nTiles = BATCH * NHEAD * (SEQ / 16);   // 4096 wave-tiles, 8 waves/block
  flash_attn<<<nTiles / 8, 256, 0, stream>>>(qb, kbuf, vt, ob);

  gemm_bf16_nt<float><<<gg, 256, 0, stream>>>(ob, wob, out, M, N, K);
}
